// TensorProductExpansion_68882685493781
// MI455X (gfx1250) — matrix operand order repaired and checked by reading, not compiled, not run on hardware
//
#include <hip/hip_runtime.h>

// TensorProductExpansion on MI455X (gfx1250, wave32).
// Strategy: bf16 hi/lo split-GEMM (3 x v_wmma_f32_16x16x32_bf16 per K-step)
// for fp32-class accuracy at bf16 matrix-pipe rate; memory-bound by design.
//
// d_ws usage: 256 KB of bf16 weight tiles (W0hi, W0lo, W1hi, W1lo), laid out
// in a contiguous-K B-matrix (32x16, 16-bit) per-lane fragment order so each lane's
// v16bf fragment is one contiguous 32-byte read.

typedef __attribute__((ext_vector_type(16))) __bf16 v16bf;
typedef __attribute__((ext_vector_type(8)))  float  v8f;

#define INV_SQRT3_F 0.57735026918962576451f
#define INV_SQRT_FAN_F 0.0625f  // 1/sqrt(2*128) = 1/16

__device__ __forceinline__ unsigned short f32_to_bf16_rn(float x) {
  unsigned int b = __float_as_uint(x);
  unsigned int r = b + 0x7FFFu + ((b >> 16) & 1u);  // round-to-nearest-even
  return (unsigned short)(r >> 16);
}
__device__ __forceinline__ float bf16_hi_to_f32(unsigned short h) {
  return __uint_as_float(((unsigned int)h) << 16);
}
__device__ __forceinline__ void split_bf16(float x, unsigned short& hi, unsigned short& lo) {
  hi = f32_to_bf16_rn(x);
  lo = f32_to_bf16_rn(x - bf16_hi_to_f32(hi));
}

// ---------------------------------------------------------------------------
// Kernel 1: convert W0/W1 (256x128 fp32, row-major, K-major rows) into bf16
// hi/lo tiles in d_ws, in WMMA B-fragment order.
// Index layout (ushort elements):
//   idx = mat*32768 + kt*4096 + nt*512 + lane*16 + e
//   mat: 0=W0hi 1=W0lo 2=W1hi 3=W1lo ; kt: K-tile (32 K per tile) ; nt: N-tile
// B-matrix (32x16, 16-bit) lane mapping used here (contiguous K, not ISA table order):
//   N = lane&15 ; K = kt*32 + (lane>>4)*16 + e
// ---------------------------------------------------------------------------
__global__ __launch_bounds__(256) void prep_weights_kernel(
    const float* __restrict__ W0, const float* __restrict__ W1,
    unsigned short* __restrict__ wsb)
{
  const int idx = blockIdx.x * 256 + threadIdx.x;   // 0 .. 131071
  if (idx >= 4 * 8 * 8 * 32 * 16) return;
  const int e    = idx & 15;
  const int lane = (idx >> 4) & 31;
  const int nt   = (idx >> 9) & 7;
  const int kt   = (idx >> 12) & 7;
  const int mat  = (idx >> 15) & 3;

  const int kp   = ((lane >> 4) << 4) | e;          // 0..31 within K-tile
  const int k    = kt * 32 + kp;                    // 0..255
  const int ncol = nt * 16 + (lane & 15);           // 0..127

  const float* W = (mat < 2) ? W0 : W1;
  const float x  = W[k * 128 + ncol];

  unsigned short hi, lo;
  split_bf16(x, hi, lo);
  wsb[idx] = (mat & 1) ? lo : hi;
}

// ---------------------------------------------------------------------------
// Kernel 2: main fused kernel. 256 threads (8 waves), 32 rows per block.
// LDS A buffer (dynamic, 128 KB):
//   A_lds[rt(2)][var(4)][h(2)][kt(8)][lane(32)][e(16)]  (ushort)
//   var: 0=scal, 1..3=vec_m ; h: 0=hi 1=lo
// A-matrix (16x32, 16-bit) lane mapping used here (contiguous K, not ISA table order):
//   M = lane&15 ; K = ((lane>>4)<<4) + e
// ---------------------------------------------------------------------------
__global__ __launch_bounds__(256) void tp_main_kernel(
    const float* __restrict__ left,  const float* __restrict__ right,
    const float* __restrict__ w_ss,  const float* __restrict__ w_sv,
    const float* __restrict__ w_vs,  const float* __restrict__ w_vv,
    const unsigned short* __restrict__ wsb,
    float* __restrict__ out, int N)
{
  extern __shared__ unsigned short A_lds[];
  const int tid = threadIdx.x;
  const int n0  = blockIdx.x * 32;

  // ------------------------- Stage 1: build A in LDS -----------------------
  {
    const int k     = tid;                 // 0..255 : K index of scal/vec
    const int kt    = k >> 5;
    const int kp    = k & 31;
    const int ahalf = kp >> 4;             // lanes 16-31 hold K+16
    const int ae    = kp & 15;

    for (int row = 0; row < 32; ++row) {
      int n = n0 + row;
      if (n >= N) n = N - 1;                       // safe clamp (N%32==0 normally)
      const float s2  = right[4 * n + 0];          // row is wave-uniform -> SMEM
      const float v2x = right[4 * n + 1];
      const float v2y = right[4 * n + 2];
      const float v2z = right[4 * n + 3];

      float scal, vx, vy, vz;
      if (k < 128) {
        const float s1 = left[(size_t)n * 512 + k];
        scal = s1 * s2 * w_ss[k];
        const float wsv = w_sv[k];
        vx = s1 * v2x * wsv;  vy = s1 * v2y * wsv;  vz = s1 * v2z * wsv;
      } else {
        const int u = k - 128;
        const float* vp = left + (size_t)n * 512 + 128 + 3 * u;
        const float a = vp[0], b = vp[1], c = vp[2];
        scal = (a * v2x + b * v2y + c * v2z) * (w_vv[u] * INV_SQRT3_F);
        const float wvs = w_vs[u] * s2;
        vx = a * wvs;  vy = b * wvs;  vz = c * wvs;
      }

      const int rt   = row >> 4;
      const int lane = (ahalf << 4) | (row & 15);
      const float vals[4] = {scal, vx, vy, vz};
      #pragma unroll
      for (int var = 0; var < 4; ++var) {
        unsigned short hi, lo;
        split_bf16(vals[var], hi, lo);
        const unsigned int base =
            ((((unsigned)(rt * 4 + var) * 2 + 0) * 8 + kt) * 32 + lane) * 16 + ae;
        A_lds[base]        = hi;   // h = 0 plane
        A_lds[base + 4096] = lo;   // h = 1 plane (stride 8*32*16 elements)
      }
    }
  }
  __syncthreads();

  // ------------------------- Stage 2: WMMA GEMMs ---------------------------
  const int lane = tid & 31;
  const int nt   = tid >> 5;            // each wave owns one 16-col strip

  const v8f vzero = {0.f, 0.f, 0.f, 0.f, 0.f, 0.f, 0.f, 0.f};
  v8f acc[4][2];
  #pragma unroll
  for (int v = 0; v < 4; ++v)
    #pragma unroll
    for (int r = 0; r < 2; ++r)
      acc[v][r] = vzero;

  const unsigned short* bbase = wsb + nt * 512 + lane * 16;

  #pragma unroll
  for (int kt = 0; kt < 8; ++kt) {
    // One set of B fragments per kt, reused by 2 row-tiles x 4 A-variants.
    const v16bf b0 = *(const v16bf*)(bbase + 0 * 32768 + kt * 4096);  // W0 hi
    const v16bf b1 = *(const v16bf*)(bbase + 1 * 32768 + kt * 4096);  // W0 lo
    const v16bf b2 = *(const v16bf*)(bbase + 2 * 32768 + kt * 4096);  // W1 hi
    const v16bf b3 = *(const v16bf*)(bbase + 3 * 32768 + kt * 4096);  // W1 lo

    #pragma unroll
    for (int r = 0; r < 2; ++r) {
      #pragma unroll
      for (int v = 0; v < 4; ++v) {
        const unsigned short* ap =
            A_lds + ((((unsigned)(r * 4 + v) * 2 + 0) * 8 + kt) * 32 + lane) * 16;
        const v16bf ahi = *(const v16bf*)ap;
        const v16bf alo = *(const v16bf*)(ap + 4096);
        const v16bf bh  = (v == 0) ? b0 : b2;
        const v16bf bl  = (v == 0) ? b1 : b3;
        v8f c = acc[v][r];
        // x*y ~= hi*hi + hi*lo + lo*hi  (drop lo*lo ~ 2^-32)
        c = __builtin_amdgcn_wmma_f32_16x16x32_bf16(false, ahi, false, bh,
                                                    (short)0, c, false, false);
        c = __builtin_amdgcn_wmma_f32_16x16x32_bf16(false, ahi, false, bl,
                                                    (short)0, c, false, false);
        c = __builtin_amdgcn_wmma_f32_16x16x32_bf16(false, alo, false, bh,
                                                    (short)0, c, false, false);
        acc[v][r] = c;
      }
    }
  }

  // ------------------------- Stage 3: scatter stores -----------------------
  const int colBase = nt * 16 + (lane & 15);
  const int rowOff  = (lane >> 4) << 3;     // lanes 16-31 hold M+8
  #pragma unroll
  for (int v = 0; v < 4; ++v) {
    const size_t cOff = (v == 0) ? (size_t)colBase
                                 : (size_t)(128 + colBase * 3 + (v - 1));
    #pragma unroll
    for (int r = 0; r < 2; ++r) {
      #pragma unroll
      for (int j = 0; j < 8; ++j) {
        const int rr = n0 + r * 16 + rowOff + j;
        if (rr < N) out[(size_t)rr * 512 + cOff] = acc[v][r][j] * INV_SQRT_FAN_F;
      }
    }
  }
}

// ---------------------------------------------------------------------------
extern "C" void kernel_launch(void* const* d_in, const int* in_sizes, int n_in,
                              void* d_out, int out_size, void* d_ws, size_t ws_size,
                              hipStream_t stream) {
  const float* left  = (const float*)d_in[0];
  const float* right = (const float*)d_in[1];
  const float* w_ss  = (const float*)d_in[2];
  const float* w_sv  = (const float*)d_in[3];
  const float* w_vs  = (const float*)d_in[4];
  const float* w_vv  = (const float*)d_in[5];
  const float* W0    = (const float*)d_in[6];
  const float* W1    = (const float*)d_in[7];
  float* out = (float*)d_out;
  unsigned short* wsb = (unsigned short*)d_ws;   // needs 256 KB

  const int N = in_sizes[0] / 512;               // 100000

  // 1) Split weights into bf16 hi/lo WMMA-B fragment layout (4*8*8*32*16 elems).
  prep_weights_kernel<<<dim3(512), dim3(256), 0, stream>>>(W0, W1, wsb);

  // 2) Fused expansion + 4 split-GEMMs. 32 rows/block, 128 KB dynamic LDS.
  const int nBlocks = (N + 31) / 32;
  tp_main_kernel<<<dim3(nBlocks), dim3(256), 131072, stream>>>(
      left, right, w_ss, w_sv, w_vs, w_vv, wsb, out, N);
}
